// GAT_42975442764294
// MI455X (gfx1250) — matrix operand order repaired and checked by reading, not compiled, not run on hardware
//
#include <hip/hip_runtime.h>
#include <hip/hip_bf16.h>

// ---------------- model constants ----------------
#define NN   1024
#define EE   8192
#define BB   4
#define TT   8
#define BT   32          // B*T
#define INC  64
#define HH   4
#define HIDD 64
#define FF   256         // H*HID
#define RR   32768       // N*B*T rows
#define DEG  8           // E/N incoming edges per node
#define KK   1018        // N - NCLASS + 1
#define NCLS 7
#define NEGS 0.2f
#define EPSL 1e-5f

typedef __attribute__((ext_vector_type(16))) _Float16 v16h;
typedef __attribute__((ext_vector_type(8)))  _Float16 v8h;
typedef __attribute__((ext_vector_type(4)))  _Float16 v4h;
typedef __attribute__((ext_vector_type(8)))  float    v8f;
typedef __attribute__((ext_vector_type(4)))  float    v4f;
typedef __attribute__((ext_vector_type(4)))  unsigned int v4u;

// ---------------------------------------------------------------
// 0a) pack weights fp32 -> f16 (one-time, L2-resident)
// ---------------------------------------------------------------
__global__ void k_pack_w(const float* __restrict__ W, _Float16* __restrict__ Wh, int count) {
  int idx = blockIdx.x * blockDim.x + threadIdx.x;
  if (idx < count) Wh[idx] = (_Float16)W[idx];
}

// ---------------------------------------------------------------
// 0b) pack inputs (B, IN, T, N) -> f16 A matrix (R x 64), r = n*32+b*8+t
//     iterate in input order so global reads are coalesced in n.
// ---------------------------------------------------------------
__global__ void k_pack_input(const float* __restrict__ in, _Float16* __restrict__ Ah) {
  int idx = blockIdx.x * blockDim.x + threadIdx.x;      // over B*IN*T*N = R*64
  if (idx >= RR * INC) return;
  int n = idx & (NN - 1);
  int t = (idx >> 10) & (TT - 1);
  int c = (idx >> 13) & (INC - 1);
  int b = idx >> 19;
  Ah[((size_t)(n * BT + b * TT + t)) * INC + c] = (_Float16)in[idx];
}

// ---------------------------------------------------------------
// 1) WMMA GEMM:  C[M x F] = A[M x K] * W^T,  A/W pre-packed f16 row-major
//    (W is [F x K], so B = W^T and B-fragment columns are W rows -> contiguous K).
//    Block = 8 waves, all sharing one 64-column W strip staged in LDS
//    (64*K f16 <= 32 KB). Wave w computes the 16(M) x 64(F) strip for
//    tileM = tileMbase + w with 4 accumulators sharing one A fragment:
//    per K-step = 2 x global b128 (A) + 4 x 32B ds_load (B) + 4 x WMMA.
// ---------------------------------------------------------------
__global__ void k_wmma_gemm_wt(const _Float16* __restrict__ Ah,
                               const _Float16* __restrict__ Wh,
                               float* __restrict__ C,
                               int M, int K, int F) {
  __shared__ _Float16 sW[64 * 256];              // 32 KB max (K<=256)

  int tilesF4   = F >> 6;                        // # of 64-wide strips
  int tileF4    = blockIdx.x % tilesF4;
  int tileMbase = (blockIdx.x / tilesF4) * 8;    // 8 waves -> 8 M-tiles (128 rows)

  // cooperative stage of the W strip: rows tileF4*64 .. +63 are contiguous
  {
    const v4u* s = (const v4u*)(Wh + (size_t)(tileF4 * 64) * K);
    v4u*       d = (v4u*)sW;
    int chunks   = (64 * K) >> 3;                // 8 halves per 16B chunk
    for (int i = threadIdx.x; i < chunks; i += blockDim.x) d[i] = s[i];
  }
  __syncthreads();

  int wv   = threadIdx.x >> 5;
  int lane = threadIdx.x & 31;
  int half = lane >> 4;          // 0: lanes 0-15, 1: lanes 16-31
  int l16  = lane & 15;
  int tileM = tileMbase + wv;
  if ((tileM << 4) >= M) return;

  const _Float16* Arow = Ah + (size_t)(tileM * 16 + l16) * K;   // A row M = l16
  const _Float16* L0 = sW + ( 0 + l16) * K;
  const _Float16* L1 = sW + (16 + l16) * K;
  const _Float16* L2 = sW + (32 + l16) * K;
  const _Float16* L3 = sW + (48 + l16) * K;

  v8f acc0 = {}, acc1 = {}, acc2 = {}, acc3 = {};
  for (int k0 = 0; k0 < K; k0 += 32) {
    if (k0 + 32 < K) __builtin_prefetch(Arow + k0 + 32, 0, 1);  // global_prefetch_b8
    // A 16x32 f16 fragment: a[0..15] = K k0+half*16+[0..15]
    v16h a = *(const v16h*)(Arow + k0 + half * 16);
    // B 32x16 f16 fragment from LDS: per lane col j=l16, 16 contiguous K values
    v16h b0 = *(const v16h*)(L0 + k0 + half * 16);
    v16h b1 = *(const v16h*)(L1 + k0 + half * 16);
    v16h b2 = *(const v16h*)(L2 + k0 + half * 16);
    v16h b3 = *(const v16h*)(L3 + k0 + half * 16);
    acc0 = __builtin_amdgcn_wmma_f32_16x16x32_f16(false, a, false, b0, (short)0, acc0, false, false);
    acc1 = __builtin_amdgcn_wmma_f32_16x16x32_f16(false, a, false, b1, (short)0, acc1, false, false);
    acc2 = __builtin_amdgcn_wmma_f32_16x16x32_f16(false, a, false, b2, (short)0, acc2, false, false);
    acc3 = __builtin_amdgcn_wmma_f32_16x16x32_f16(false, a, false, b3, (short)0, acc3, false, false);
  }
  // D 16x16 f32: lane col = l16; VGPR v -> row M = v + half*8
#pragma unroll
  for (int v = 0; v < 8; ++v) {
    size_t row = (size_t)(tileM * 16 + v + half * 8) * F + tileF4 * 64 + l16;
    C[row +  0] = acc0[v];
    C[row + 16] = acc1[v];
    C[row + 32] = acc2[v];
    C[row + 48] = acc3[v];
  }
}

// ---------------------------------------------------------------
// 2) attention logits: el/er[(n*32+bt)*4+h] = <feat[r, h*64:..], al/ar[h]>
// ---------------------------------------------------------------
__global__ void k_attn_logits(const float* __restrict__ feat,
                              const float* __restrict__ al,
                              const float* __restrict__ ar,
                              float* __restrict__ el, float* __restrict__ er) {
  int idx = blockIdx.x * blockDim.x + threadIdx.x;      // over N*BT*H
  if (idx >= NN * BT * HH) return;
  int h  = idx & (HH - 1);
  int r  = idx >> 2;                                    // n*32+bt
  const float* f   = feat + (size_t)r * FF + h * HIDD;
  const float* alh = al + h * HIDD;
  const float* arh = ar + h * HIDD;
  float sl = 0.f, sr = 0.f;
#pragma unroll 8
  for (int d = 0; d < HIDD; ++d) { float v = f[d]; sl += v * alh[d]; sr += v * arh[d]; }
  el[idx] = sl; er[idx] = sr;
}

// layer-2 variant: HID=1, logits are just feat2 * scalar
__global__ void k_attn_logits_l2(const float* __restrict__ feat2,
                                 const float* __restrict__ al2,
                                 const float* __restrict__ ar2,
                                 float* __restrict__ el, float* __restrict__ er) {
  int idx = blockIdx.x * blockDim.x + threadIdx.x;
  if (idx >= NN * BT * HH) return;
  int h = idx & (HH - 1);
  float v = feat2[idx];
  el[idx] = v * al2[h];
  er[idx] = v * ar2[h];
}

// ---------------------------------------------------------------
// 3) per-destination softmax over the fixed 8 incoming edges
//    attw[((n*8+j)*32+bt)*4+h]
// ---------------------------------------------------------------
__global__ void k_edge_softmax(const int* __restrict__ src,
                               const float* __restrict__ el,
                               const float* __restrict__ er,
                               float* __restrict__ attw) {
  int idx = blockIdx.x * blockDim.x + threadIdx.x;      // over N*BT*H
  if (idx >= NN * BT * HH) return;
  int h  = idx & (HH - 1);
  int bt = (idx >> 2) & (BT - 1);
  int n  = idx >> 7;
  float ern = er[idx];
  float sc[DEG];
  float mx = -1e30f;
#pragma unroll
  for (int j = 0; j < DEG; ++j) {
    int s = src[n * DEG + j];
    float e = el[((size_t)s * BT + bt) * HH + h] + ern;
    e = (e > 0.f) ? e : NEGS * e;                       // leaky_relu
    sc[j] = e;
    mx = fmaxf(mx, e);
  }
  float den = 0.f;
#pragma unroll
  for (int j = 0; j < DEG; ++j) { sc[j] = __expf(sc[j] - mx); den += sc[j]; }
  float inv = 1.f / den;
#pragma unroll
  for (int j = 0; j < DEG; ++j)
    attw[(((size_t)(n * DEG + j)) * BT + bt) * HH + h] = sc[j] * inv;
}

// ---------------------------------------------------------------
// 4) aggregate (vectorized 4 features/thread):
//    out[n,bt,f..f+3] = sum_j a[j] * feat[src_j,bt,f..f+3] (+res) (elu?)
//    optionally dual-writes f16 copy feeding the next WMMA GEMM.
// ---------------------------------------------------------------
__global__ void k_aggregate(const int* __restrict__ src,
                            const float* __restrict__ attw,
                            const float* __restrict__ feat,
                            const float* __restrict__ res,
                            float* __restrict__ out,
                            _Float16* __restrict__ out16,
                            int has_res, int do_elu, int has16) {
  int idx = blockIdx.x * blockDim.x + threadIdx.x;      // over R*64 groups of 4 feats
  if (idx >= RR * (FF / 4)) return;
  int fq = idx & (FF / 4 - 1);                          // feature group
  int f  = fq << 2;
  int bt = (idx >> 6) & (BT - 1);
  int n  = idx >> 11;
  int h  = f >> 6;
  v4f acc = {};
#pragma unroll
  for (int j = 0; j < DEG; ++j) {
    int s   = src[n * DEG + j];
    float a = attw[(((size_t)(n * DEG + j)) * BT + bt) * HH + h];
    v4f fv  = *(const v4f*)(feat + ((size_t)s * BT + bt) * FF + f);
    acc += a * fv;
  }
  size_t o = ((size_t)(n * BT + bt)) * FF + f;
  if (has_res) acc += *(const v4f*)(res + o);
  if (do_elu) {
#pragma unroll
    for (int i = 0; i < 4; ++i) acc[i] = (acc[i] > 0.f) ? acc[i] : (__expf(acc[i]) - 1.f);
  }
  *(v4f*)(out + o) = acc;
  if (has16) {
    v4h a16;
#pragma unroll
    for (int i = 0; i < 4; ++i) a16[i] = (_Float16)acc[i];
    *(v4h*)(out16 + o) = a16;
  }
}

// ---------------------------------------------------------------
// 5) layer-2 projections: feat2/res2[r,h] = <h2[r,:], W2/Wres2[h,:]>
// ---------------------------------------------------------------
__global__ void k_proj_l2(const float* __restrict__ h2,
                          const float* __restrict__ W2,
                          const float* __restrict__ Wres2,
                          float* __restrict__ feat2, float* __restrict__ res2) {
  int idx = blockIdx.x * blockDim.x + threadIdx.x;      // over R*H
  if (idx >= RR * HH) return;
  int h = idx & (HH - 1);
  int r = idx >> 2;
  const float* x  = h2 + (size_t)r * FF;
  const float* w  = W2 + h * FF;
  const float* wr = Wres2 + h * FF;
  float s = 0.f, sr = 0.f;
#pragma unroll 8
  for (int c = 0; c < FF; ++c) { float v = x[c]; s += v * w[c]; sr += v * wr[c]; }
  feat2[idx] = s; res2[idx] = sr;
}

// 6) layer-2 aggregate (HID=1), + res2, no activation
__global__ void k_aggregate_l2(const int* __restrict__ src,
                               const float* __restrict__ attw,
                               const float* __restrict__ feat2,
                               const float* __restrict__ res2,
                               float* __restrict__ o) {
  int idx = blockIdx.x * blockDim.x + threadIdx.x;      // over N*BT*H
  if (idx >= NN * BT * HH) return;
  int h  = idx & (HH - 1);
  int bt = (idx >> 2) & (BT - 1);
  int n  = idx >> 7;
  float acc = res2[idx];
#pragma unroll
  for (int j = 0; j < DEG; ++j) {
    int s = src[n * DEG + j];
    acc += attw[(((size_t)(n * DEG + j)) * BT + bt) * HH + h]
         * feat2[((size_t)s * BT + bt) * HH + h];
  }
  o[idx] = acc;
}

// ---------------------------------------------------------------
// 7) temporal conv 1: x1[b,o,n] = sum_{h,t} tc1_w[o,h,t] * o[n,b,t,h] + tc1_b[o]
// ---------------------------------------------------------------
__global__ void k_tc1(const float* __restrict__ o,
                      const float* __restrict__ w,
                      const float* __restrict__ bias,
                      float* __restrict__ x1) {
  int idx = blockIdx.x * blockDim.x + threadIdx.x;      // over B*4*N
  if (idx >= BB * HH * NN) return;
  int n  = idx & (NN - 1);
  int oo = (idx >> 10) & (HH - 1);
  int b  = idx >> 12;
  float s = bias[oo];
#pragma unroll
  for (int h = 0; h < HH; ++h)
#pragma unroll
    for (int t = 0; t < TT; ++t)
      s += w[(oo * HH + h) * TT + t] * o[((size_t)(n * BT + b * TT + t)) * HH + h];
  x1[((size_t)b * HH + oo) * NN + n] = s;
}

// 8) layernorm over (N,H) per batch; x1 layout [b][h][n], gamma/beta (N,H)
__global__ void k_ln1(const float* __restrict__ x1,
                      const float* __restrict__ g, const float* __restrict__ bb,
                      float* __restrict__ x1n) {
  int b   = blockIdx.x;
  int tid = threadIdx.x;
  __shared__ float ssum[256], ssq[256];
  float s = 0.f, sq = 0.f;
  for (int i = tid; i < HH * NN; i += 256) {
    float v = x1[b * HH * NN + i]; s += v; sq += v * v;
  }
  ssum[tid] = s; ssq[tid] = sq; __syncthreads();
  for (int st = 128; st > 0; st >>= 1) {
    if (tid < st) { ssum[tid] += ssum[tid + st]; ssq[tid] += ssq[tid + st]; }
    __syncthreads();
  }
  float mu  = ssum[0] / (float)(HH * NN);
  float var = ssq[0] / (float)(HH * NN) - mu * mu;
  float inv = rsqrtf(var + EPSL);
  for (int i = tid; i < HH * NN; i += 256) {
    int h = i >> 10, n = i & (NN - 1);
    float v = (x1[b * HH * NN + i] - mu) * inv;
    x1n[b * HH * NN + i] = v * g[n * HH + h] + bb[n * HH + h];
  }
}

// 9) temporal conv 2: x2[b,n] = sum_h tc2_w[h] * x1n[b,h,n] + tc2_b
__global__ void k_tc2(const float* __restrict__ x1n,
                      const float* __restrict__ w, const float* __restrict__ bias,
                      float* __restrict__ x2) {
  int idx = blockIdx.x * blockDim.x + threadIdx.x;      // over B*N
  if (idx >= BB * NN) return;
  int n = idx & (NN - 1);
  int b = idx >> 10;
  float s = bias[0];
#pragma unroll
  for (int h = 0; h < HH; ++h) s += w[h] * x1n[((size_t)b * HH + h) * NN + n];
  x2[idx] = s;
}

// 10) layernorm over N per batch; gamma/beta from (N,1)
__global__ void k_ln2(const float* __restrict__ x2,
                      const float* __restrict__ g, const float* __restrict__ bb,
                      float* __restrict__ x2n) {
  int b   = blockIdx.x;
  int tid = threadIdx.x;
  __shared__ float ssum[256], ssq[256];
  float s = 0.f, sq = 0.f;
  for (int i = tid; i < NN; i += 256) { float v = x2[b * NN + i]; s += v; sq += v * v; }
  ssum[tid] = s; ssq[tid] = sq; __syncthreads();
  for (int st = 128; st > 0; st >>= 1) {
    if (tid < st) { ssum[tid] += ssum[tid + st]; ssq[tid] += ssq[tid + st]; }
    __syncthreads();
  }
  float mu  = ssum[0] / (float)NN;
  float var = ssq[0] / (float)NN - mu * mu;
  float inv = rsqrtf(var + EPSL);
  for (int i = tid; i < NN; i += 256)
    x2n[b * NN + i] = (x2[b * NN + i] - mu) * inv * g[i] + bb[i];
}

// 11) head conv: out[b,k] = sum_i x2n[b,k+i]*fc_w[i] + fc_b, k in [0,7)
__global__ void k_head(const float* __restrict__ x2n,
                       const float* __restrict__ fcw, const float* __restrict__ fcb,
                       float* __restrict__ out) {
  int b = blockIdx.x / NCLS;
  int k = blockIdx.x % NCLS;
  int tid = threadIdx.x;
  __shared__ float ss[256];
  float s = 0.f;
  for (int i = tid; i < KK; i += 256) s += x2n[b * NN + k + i] * fcw[i];
  ss[tid] = s; __syncthreads();
  for (int st = 128; st > 0; st >>= 1) {
    if (tid < st) ss[tid] += ss[tid + st];
    __syncthreads();
  }
  if (tid == 0) out[b * NCLS + k] = ss[0] + fcb[0];
}

// ---------------------------------------------------------------
extern "C" void kernel_launch(void* const* d_in, const int* in_sizes, int n_in,
                              void* d_out, int out_size, void* d_ws, size_t ws_size,
                              hipStream_t stream) {
  const float* inputs = (const float*)d_in[0];
  const int*   src    = (const int*)d_in[1];
  // d_in[2] = dst (deterministic repeat pattern, unused)
  const float* W0    = (const float*)d_in[3];
  const float* al0   = (const float*)d_in[4];
  const float* ar0   = (const float*)d_in[5];
  const float* W1    = (const float*)d_in[6];
  const float* al1   = (const float*)d_in[7];
  const float* ar1   = (const float*)d_in[8];
  const float* W2    = (const float*)d_in[9];
  const float* al2   = (const float*)d_in[10];
  const float* ar2   = (const float*)d_in[11];
  const float* Wres2 = (const float*)d_in[12];
  const float* tc1w  = (const float*)d_in[13];
  const float* tc1b  = (const float*)d_in[14];
  const float* ln1g  = (const float*)d_in[15];
  const float* ln1b  = (const float*)d_in[16];
  const float* tc2w  = (const float*)d_in[17];
  const float* tc2b  = (const float*)d_in[18];
  const float* ln2g  = (const float*)d_in[19];
  const float* ln2b  = (const float*)d_in[20];
  const float* fcw   = (const float*)d_in[21];
  const float* fcb   = (const float*)d_in[22];
  float* out = (float*)d_out;

  // ---- workspace carve-up ----
  // f16 region first (ws base is allocation-aligned; keeps b128 fragment loads aligned)
  _Float16* Ah  = (_Float16*)d_ws;                   // R*FF halves (16 MB); layer0 uses R x 64 view
  _Float16* Wh0 = Ah  + (size_t)RR * FF;             // 256*64
  _Float16* Wh1 = Wh0 + (size_t)FF * INC;            // 256*256
  float* fbase  = (float*)(Wh1 + (size_t)FF * FF);
  float* feat = fbase;                               // R*FF (32 MB)
  float* h1   = feat + (size_t)RR * FF;              // R*FF (32 MB)
  float* h2   = h1   + (size_t)RR * FF;              // R*FF (32 MB)
  float* el   = h2   + (size_t)RR * FF;              // N*BT*H
  float* er   = el   + (size_t)NN * BT * HH;
  float* attw = er   + (size_t)NN * BT * HH;         // E*BT*H (4 MB)
  float* feat2= attw + (size_t)EE * BT * HH;         // N*BT*H
  float* res2 = feat2+ (size_t)NN * BT * HH;
  float* obuf = res2 + (size_t)NN * BT * HH;
  float* x1   = obuf + (size_t)NN * BT * HH;         // B*H*N
  float* x1n  = x1   + (size_t)BB * HH * NN;
  float* x2   = x1n  + (size_t)BB * HH * NN;         // B*N
  float* x2n  = x2   + (size_t)BB * NN;

  const int TPB  = 256;
  const int gNBH = (NN * BT * HH) / TPB;             // 512
  const int gAgg = (RR * (FF / 4)) / TPB;            // 8192 (4 feats/thread)
  const int gPack= (RR * INC) / TPB;                 // 8192
  // GEMM: 8 waves/block, block covers 128 M-rows x 64 F-cols
  const int gemmBlocks = (RR / 128) * (FF / 64);     // 1024

  // ---- one-time weight packs ----
  k_pack_w<<<(FF * INC) / TPB, TPB, 0, stream>>>(W0, Wh0, FF * INC);
  k_pack_w<<<(FF * FF) / TPB, TPB, 0, stream>>>(W1, Wh1, FF * FF);

  // ---- layer 0 ----
  k_pack_input<<<gPack, TPB, 0, stream>>>(inputs, Ah);   // R x 64 f16 view
  k_wmma_gemm_wt<<<gemmBlocks, TPB, 0, stream>>>(Ah, Wh0, feat, RR, INC, FF);
  k_attn_logits<<<gNBH, TPB, 0, stream>>>(feat, al0, ar0, el, er);
  k_edge_softmax<<<gNBH, TPB, 0, stream>>>(src, el, er, attw);
  // h1 fp32 for residual + f16 copy into Ah (R x 256) for the next GEMM
  k_aggregate<<<gAgg, TPB, 0, stream>>>(src, attw, feat, nullptr, h1, Ah, 0, 1, 1);

  // ---- layer 1 ----
  k_wmma_gemm_wt<<<gemmBlocks, TPB, 0, stream>>>(Ah, Wh1, feat, RR, FF, FF);
  k_attn_logits<<<gNBH, TPB, 0, stream>>>(feat, al1, ar1, el, er);
  k_edge_softmax<<<gNBH, TPB, 0, stream>>>(src, el, er, attw);
  k_aggregate<<<gAgg, TPB, 0, stream>>>(src, attw, feat, h1, h2, nullptr, 1, 1, 0);

  // ---- layer 2 (H outputs, HID=1) ----
  k_proj_l2<<<gNBH, TPB, 0, stream>>>(h2, W2, Wres2, feat2, res2);
  k_attn_logits_l2<<<gNBH, TPB, 0, stream>>>(feat2, al2, ar2, el, er);
  k_edge_softmax<<<gNBH, TPB, 0, stream>>>(src, el, er, attw);
  k_aggregate_l2<<<gNBH, TPB, 0, stream>>>(src, attw, feat2, res2, obuf);

  // ---- temporal head ----
  k_tc1<<<(BB * HH * NN) / TPB, TPB, 0, stream>>>(obuf, tc1w, tc1b, x1);
  k_ln1<<<BB, TPB, 0, stream>>>(x1, ln1g, ln1b, x1n);
  k_tc2<<<(BB * NN) / TPB, TPB, 0, stream>>>(x1n, tc2w, tc2b, x2);
  k_ln2<<<BB, TPB, 0, stream>>>(x2, ln2g, ln2b, x2n);
  k_head<<<BB * NCLS, TPB, 0, stream>>>(x2n, fcw, fcb, out);
}
